// DeepseekMoE_1297080123443
// MI455X (gfx1250) — matrix operand order repaired and checked by reading, not compiled, not run on hardware
//
#include <hip/hip_runtime.h>
#include <hip/hip_bf16.h>
#include <stdint.h>

// ---------------- problem constants ----------------
#define TOKENS      4096
#define NUM_EXPERTS 16
#define TOP_K       6
#define HIDDEN      2048
#define INTER       1408
#define NBLK_H      16
#define NBLK_I      11
#define PAIRS       (TOKENS * TOP_K)   // 24576

// ---------------- tiling ----------------
#define TILE_T      64
#define ACT_STRIDE  (INTER + 16)       // padded bf16 row stride (bank stagger)
#define WAVES       8
#define THREADS     (WAVES * 32)
#define MAX_TILES   (PAIRS / TILE_T)   // 384

typedef __bf16 bf16;
typedef __attribute__((ext_vector_type(16))) __bf16 v16bf;
typedef __attribute__((ext_vector_type(8)))  float  v8f;
typedef __attribute__((ext_vector_type(4)))  float  v4f;
typedef __attribute__((ext_vector_type(4)))  unsigned int v4u;

// ---------------- workspace layout ----------------
constexpr size_t OFF_LISTS   = 256;
constexpr size_t LISTS_BYTES = (size_t)NUM_EXPERTS * PAIRS * 4;        // 1.5 MB
constexpr size_t OFF_XBF     = OFF_LISTS + LISTS_BYTES;
constexpr size_t XBF_BYTES   = (size_t)TOKENS * HIDDEN * 2;            // 16 MB
constexpr size_t OFF_W0BF    = OFF_XBF + XBF_BYTES;
constexpr size_t WBF_BYTES   = (size_t)NUM_EXPERTS * INTER * HIDDEN * 2; // 92 MB
constexpr size_t OFF_W1BF    = OFF_W0BF + WBF_BYTES;
constexpr size_t OFF_W2BF    = OFF_W1BF + WBF_BYTES;
constexpr size_t WS_NEEDED   = OFF_W2BF + WBF_BYTES;                   // ~295 MB

__device__ __forceinline__ v8f vzero8() {
    v8f z;
#pragma unroll
    for (int j = 0; j < 8; ++j) z[j] = 0.0f;
    return z;
}

// ---- bf16 fragment loaders (pure b128 loads, no VALU) ----
// A (16x32): lane M = ln, 16 contiguous K starting at 16*hi
__device__ __forceinline__ v16bf make_a_bf16(const bf16* rowp, int kc, int hi) {
    union { v16bf v; v4u q[2]; } u;
    const bf16* p = rowp + kc + (hi << 4);
    u.q[0] = *(const v4u*)(p);
    u.q[1] = *(const v4u*)(p + 8);
    return u.v;
}
// B (32x16): lane N = ln, 16 contiguous K starting at 16*hi
__device__ __forceinline__ v16bf make_b_bf16(const bf16* rowp, int kc, int hi) {
    union { v16bf v; v4u q[2]; } u;
    const bf16* p = rowp + kc + (hi << 4);
    u.q[0] = *(const v4u*)(p);
    u.q[1] = *(const v4u*)(p + 8);
    return u.v;
}

// ---- f32 fragment builders with fused dequant (fallback path) ----
__device__ __forceinline__ v16bf make_a_f32(const float* __restrict__ rowp, int kc, int hi) {
    const float* p = rowp + kc + (hi << 4);
    v4f f0 = *(const v4f*)(p);
    v4f f1 = *(const v4f*)(p + 4);
    v4f f2 = *(const v4f*)(p + 8);
    v4f f3 = *(const v4f*)(p + 12);
    v16bf a;
#pragma unroll
    for (int j = 0; j < 4; ++j) {
        a[j] = (bf16)f0[j]; a[4 + j] = (bf16)f1[j];
        a[8 + j] = (bf16)f2[j]; a[12 + j] = (bf16)f3[j];
    }
    return a;
}
__device__ __forceinline__ v16bf make_b_f32(const float* __restrict__ rowp, int kc, int hi, float s) {
    const float* p = rowp + kc + (hi << 4);
    v4f f0 = *(const v4f*)(p);
    v4f f1 = *(const v4f*)(p + 4);
    v4f f2 = *(const v4f*)(p + 8);
    v4f f3 = *(const v4f*)(p + 12);
    v16bf b;
#pragma unroll
    for (int j = 0; j < 4; ++j) {
        b[j] = (bf16)(f0[j] * s); b[4 + j] = (bf16)(f1[j] * s);
        b[8 + j] = (bf16)(f2[j] * s); b[12 + j] = (bf16)(f3[j] * s);
    }
    return b;
}

#define WMMA_BF16(a, b, c) \
    __builtin_amdgcn_wmma_f32_16x16x32_bf16(false, (a), false, (b), (short)0, (c), false, false)

// ---------------- routing ----------------
__global__ void k_zero_counts(int* counts) {
    if (threadIdx.x < NUM_EXPERTS) counts[threadIdx.x] = 0;
}

__global__ void k_route(const int* __restrict__ sel, int* counts, int* lists) {
    int idx = blockIdx.x * blockDim.x + threadIdx.x;   // token*TOP_K + slot
    if (idx >= PAIRS) return;
    int e = sel[idx] & (NUM_EXPERTS - 1);
    int pos = atomicAdd(&counts[e], 1);
    lists[e * PAIRS + pos] = idx;
}

// ---------------- pre-conversion kernels ----------------
__global__ void k_cvt_plain(const float* __restrict__ src, bf16* __restrict__ dst, int n8) {
    int i = blockIdx.x * blockDim.x + threadIdx.x;
    if (i >= n8) return;
    int i8 = i * 8;
    v4f f0 = *(const v4f*)(src + i8);
    v4f f1 = *(const v4f*)(src + i8 + 4);
    union { v4u q; bf16 e[8]; } u;
#pragma unroll
    for (int j = 0; j < 4; ++j) { u.e[j] = (bf16)f0[j]; u.e[4 + j] = (bf16)f1[j]; }
    *(v4u*)(dst + i8) = u.q;
}

template <int K, int NBLK>
__global__ void k_cvt_scaled(const float* __restrict__ w, const float* __restrict__ s,
                             bf16* __restrict__ dst, int n8) {
    int i = blockIdx.x * blockDim.x + threadIdx.x;
    if (i >= n8) return;
    int i8 = i * 8;                       // 8 contiguous elems, never cross a 128 block
    int row = i8 / K;
    int col = i8 - row * K;
    float sc = s[row * NBLK + (col >> 7)];
    v4f f0 = *(const v4f*)(w + i8);
    v4f f1 = *(const v4f*)(w + i8 + 4);
    union { v4u q; bf16 e[8]; } u;
#pragma unroll
    for (int j = 0; j < 4; ++j) { u.e[j] = (bf16)(f0[j] * sc); u.e[4 + j] = (bf16)(f1[j] * sc); }
    *(v4u*)(dst + i8) = u.q;
}

// ---------------- fused MoE expert kernel ----------------
// grid = (MAX_TILES, NUM_EXPERTS), 8 waves.
// wave w: token blocks {m2, m2+1}, m2 = (w&1)*2 ; column quad nq = w>>1 (0..3)
template <bool PRE>
__global__ __launch_bounds__(THREADS, 1)
void moe_fused(const float* __restrict__ x,
               const float* __restrict__ w0, const float* __restrict__ s0,
               const float* __restrict__ w1, const float* __restrict__ s1,
               const float* __restrict__ w2, const float* __restrict__ s2,
               const bf16* __restrict__ xbf,
               const bf16* __restrict__ w0bf,
               const bf16* __restrict__ w1bf,
               const bf16* __restrict__ w2bf,
               const int* __restrict__ counts,
               const int* __restrict__ lists,
               float* __restrict__ out)
{
    __shared__ bf16 sAct[TILE_T * ACT_STRIDE];
    __shared__ int  sPair[TILE_T];

    const int e    = blockIdx.y;
    const int cnt  = counts[e];
    const int base = blockIdx.x * TILE_T;
    if (base >= cnt) return;

    const int tid  = threadIdx.x;
    const int lane = tid & 31;
    const int wave = tid >> 5;
    const int ln   = lane & 15;
    const int hi   = lane >> 4;

    const int* mylist = lists + e * PAIRS;
    if (tid < TILE_T) {
        int r = base + tid;
        sPair[tid] = mylist[(r < cnt) ? r : base];
    }
    __syncthreads();

    const int m2 = (wave & 1) * 2;   // token-block pair base
    const int nq = wave >> 1;        // column quad 0..3

    int rowA[2];
    const float* xrow[2];
    const bf16*  xrowb[2];
#pragma unroll
    for (int mi = 0; mi < 2; ++mi) {
        rowA[mi] = (m2 + mi) * 16 + ln;
        int tok = sPair[rowA[mi]] / TOP_K;
        xrow[mi]  = x   + (size_t)tok * HIDDEN;
        xrowb[mi] = xbf + (size_t)tok * HIDDEN;
    }

    // ============ Pass 1: act = silu(x@W0^T) * (x@W1^T) -> LDS ============
    for (int ic = 0; ic < INTER / 128; ++ic) {
        const bf16 *b0b[2], *b1b[2];
        const float *b0f[2], *b1f[2], *s0p[2], *s1p[2];
#pragma unroll
        for (int t = 0; t < 2; ++t) {
            int i = ic * 128 + nq * 32 + t * 16 + ln;
            size_t wr = (size_t)e * INTER + i;
            if constexpr (PRE) {
                b0b[t] = w0bf + wr * HIDDEN;
                b1b[t] = w1bf + wr * HIDDEN;
            } else {
                b0f[t] = w0 + wr * HIDDEN;  b1f[t] = w1 + wr * HIDDEN;
                s0p[t] = s0 + wr * NBLK_H;  s1p[t] = s1 + wr * NBLK_H;
            }
        }
        v8f cg[2][2], cu[2][2];
#pragma unroll
        for (int mi = 0; mi < 2; ++mi)
#pragma unroll
            for (int t = 0; t < 2; ++t) { cg[mi][t] = vzero8(); cu[mi][t] = vzero8(); }

        for (int kb = 0; kb < NBLK_H; ++kb) {
            float sc0[2], sc1[2];
            if constexpr (!PRE) {
#pragma unroll
                for (int t = 0; t < 2; ++t) { sc0[t] = s0p[t][kb]; sc1[t] = s1p[t][kb]; }
            }
#pragma unroll
            for (int kk = 0; kk < 4; ++kk) {
                int kc = kb * 128 + kk * 32;
                v16bf a[2];
#pragma unroll
                for (int mi = 0; mi < 2; ++mi)
                    a[mi] = PRE ? make_a_bf16(xrowb[mi], kc, hi)
                                : make_a_f32(xrow[mi], kc, hi);
#pragma unroll
                for (int t = 0; t < 2; ++t) {
                    v16bf b0 = PRE ? make_b_bf16(b0b[t], kc, hi)
                                   : make_b_f32(b0f[t], kc, hi, sc0[t]);
#pragma unroll
                    for (int mi = 0; mi < 2; ++mi) cg[mi][t] = WMMA_BF16(a[mi], b0, cg[mi][t]);
                    v16bf b1 = PRE ? make_b_bf16(b1b[t], kc, hi)
                                   : make_b_f32(b1f[t], kc, hi, sc1[t]);
#pragma unroll
                    for (int mi = 0; mi < 2; ++mi) cu[mi][t] = WMMA_BF16(a[mi], b1, cu[mi][t]);
                }
            }
        }
#pragma unroll
        for (int mi = 0; mi < 2; ++mi)
#pragma unroll
            for (int t = 0; t < 2; ++t) {
                int colb = ic * 128 + nq * 32 + t * 16 + ln;
#pragma unroll
                for (int j = 0; j < 8; ++j) {
                    int row = (m2 + mi) * 16 + j + hi * 8;
                    float g = cg[mi][t][j], u = cu[mi][t][j];
                    float av = (g / (1.0f + __expf(-g))) * u;
                    sAct[row * ACT_STRIDE + colb] = (bf16)av;
                }
            }
    }
    __syncthreads();

    // ============ Pass 2: out = act @ W2^T ============
    const bf16* aR[2] = { sAct + (size_t)rowA[0] * ACT_STRIDE,
                          sAct + (size_t)rowA[1] * ACT_STRIDE };

    for (int hb = 0; hb < HIDDEN / 256; ++hb) {
        const bf16* b2b[4];
        const float *b2f[4], *s2p[4];
#pragma unroll
        for (int t = 0; t < 4; ++t) {
            int h = hb * 256 + nq * 64 + t * 16 + ln;
            size_t wr = (size_t)e * HIDDEN + h;
            if constexpr (PRE) b2b[t] = w2bf + wr * INTER;
            else { b2f[t] = w2 + wr * INTER; s2p[t] = s2 + wr * NBLK_I; }
        }
        v8f co[2][4];
#pragma unroll
        for (int mi = 0; mi < 2; ++mi)
#pragma unroll
            for (int t = 0; t < 4; ++t) co[mi][t] = vzero8();

        for (int kb = 0; kb < NBLK_I; ++kb) {
            float sc[4];
            if constexpr (!PRE) {
#pragma unroll
                for (int t = 0; t < 4; ++t) sc[t] = s2p[t][kb];
            }
#pragma unroll
            for (int kk = 0; kk < 4; ++kk) {
                int kc = kb * 128 + kk * 32;
                v16bf a[2];
#pragma unroll
                for (int mi = 0; mi < 2; ++mi) a[mi] = make_a_bf16(aR[mi], kc, hi);
#pragma unroll
                for (int t = 0; t < 4; ++t) {
                    v16bf b = PRE ? make_b_bf16(b2b[t], kc, hi)
                                  : make_b_f32(b2f[t], kc, hi, sc[t]);
#pragma unroll
                    for (int mi = 0; mi < 2; ++mi) co[mi][t] = WMMA_BF16(a[mi], b, co[mi][t]);
                }
            }
        }
#pragma unroll
        for (int mi = 0; mi < 2; ++mi)
#pragma unroll
            for (int j = 0; j < 8; ++j) {
                int row = (m2 + mi) * 16 + j + hi * 8;
                size_t obase = (size_t)sPair[row] * HIDDEN;
#pragma unroll
                for (int t = 0; t < 4; ++t) {
                    int hcol = hb * 256 + nq * 64 + t * 16 + ln;
                    out[obase + hcol] = co[mi][t][j];
                }
            }
    }
}

// ---------------- launcher ----------------
extern "C" void kernel_launch(void* const* d_in, const int* in_sizes, int n_in,
                              void* d_out, int out_size, void* d_ws, size_t ws_size,
                              hipStream_t stream) {
    const float* x   = (const float*)d_in[0];
    const int*   sel = (const int*)  d_in[1];
    const float* w0  = (const float*)d_in[2];
    const float* s0  = (const float*)d_in[3];
    const float* w1  = (const float*)d_in[4];
    const float* s1  = (const float*)d_in[5];
    const float* w2  = (const float*)d_in[6];
    const float* s2  = (const float*)d_in[7];
    float* out = (float*)d_out;

    char* ws = (char*)d_ws;
    int* counts = (int*)ws;
    int* lists  = (int*)(ws + OFF_LISTS);

    k_zero_counts<<<dim3(1), dim3(32), 0, stream>>>(counts);
    k_route<<<dim3((PAIRS + 255) / 256), dim3(256), 0, stream>>>(sel, counts, lists);

    const dim3 grid(MAX_TILES, NUM_EXPERTS);
    const dim3 blk(THREADS);

    if (ws_size >= WS_NEEDED) {
        bf16* xbf  = (bf16*)(ws + OFF_XBF);
        bf16* w0bf = (bf16*)(ws + OFF_W0BF);
        bf16* w1bf = (bf16*)(ws + OFF_W1BF);
        bf16* w2bf = (bf16*)(ws + OFF_W2BF);

        const int n8x = TOKENS * HIDDEN / 8;               // 1,048,576
        const int n8w = NUM_EXPERTS * INTER * HIDDEN / 8;  // 5,767,168
        k_cvt_plain<<<dim3((n8x + 255) / 256), dim3(256), 0, stream>>>(x, xbf, n8x);
        k_cvt_scaled<HIDDEN, NBLK_H><<<dim3((n8w + 255) / 256), dim3(256), 0, stream>>>(w0, s0, w0bf, n8w);
        k_cvt_scaled<HIDDEN, NBLK_H><<<dim3((n8w + 255) / 256), dim3(256), 0, stream>>>(w1, s1, w1bf, n8w);
        k_cvt_scaled<INTER, NBLK_I><<<dim3((n8w + 255) / 256), dim3(256), 0, stream>>>(w2, s2, w2bf, n8w);

        moe_fused<true><<<grid, blk, 0, stream>>>(x, w0, s0, w1, s1, w2, s2,
                                                  xbf, w0bf, w1bf, w2bf,
                                                  counts, lists, out);
    } else {
        moe_fused<false><<<grid, blk, 0, stream>>>(x, w0, s0, w1, s1, w2, s2,
                                                   nullptr, nullptr, nullptr, nullptr,
                                                   counts, lists, out);
    }
}
